// SingleScaleAlignment_47863115546788
// MI455X (gfx1250) — compile-verified, matrix operand order checked by reading, not run on hardware
//
#include <hip/hip_runtime.h>
#include <hip/hip_bf16.h>

// ---------------------------------------------------------------------------
// Deformable-conv temporal alignment (EDVR-style, single scale), CDNA5 WMMA.
// B=1, T=5, C=64, H=W=96, G=8, Cg=8, K=3 (KK=9), PAD=1.
// Per non-center frame:
//   GEMM1 (offset conv): M=9216, N=144, K=1152  -> v_wmma_f32_16x16x32_bf16
//                        3 N-tiles per wave (A-fragment reuse x3)
//   bilinear deformable sampling (VALU)         -> bf16 col buffer [9216][576]
//   GEMM2 (dcn reduce):  M=9216, N=64,  K=576   -> v_wmma_f32_16x16x32_bf16
//                        4 N-tiles per wave (A-fragment reuse x4)
// ---------------------------------------------------------------------------

typedef __attribute__((ext_vector_type(16))) __bf16 v16bf;
typedef __attribute__((ext_vector_type(8)))  float  v8f;

union FragBF {
    v16bf v;
    uint4 q[2];
};

static __device__ __forceinline__ unsigned short f2bf(float f) {
    unsigned int u = __float_as_uint(f);
    unsigned int r = (u + 0x7FFFu + ((u >> 16) & 1u)) >> 16;   // RNE
    return (unsigned short)r;
}

// -------------------- constants --------------------
#define HH   96
#define WW   96
#define HWC  9216          // 96*96
#define CIN  64
#define C2   128           // concat channels
#define NOFF 144           // offset channels = G*2*KK
#define K1   1152          // C2*9
#define K2   576           // G*Cg*KK
#define PADW 98            // 96+2

// ===========================================================================
// Weight swizzle into WMMA B-fragment order: [ktile][ntile][lane][16] bf16.
// B element (k, n): lane l -> n = ntile*16 + (l&15);
//   e<8: k = kt*32 + 8*(l>>4) + e ; e>=8: k = kt*32 + 16 + 8*(l>>4) + (e-8)
// ===========================================================================
__global__ void swz_woff(const float* __restrict__ w, unsigned short* __restrict__ out) {
    int tid = blockIdx.x * blockDim.x + threadIdx.x;      // 36*9*32*16
    int e = tid & 15;
    int l = (tid >> 4) & 31;
    int rest = tid >> 9;
    int nt = rest % 9;
    int s  = rest / 9;                                    // ktile 0..35
    int half = l >> 4;
    int kit = (e < 8) ? (8 * half + e) : (16 + 8 * half + (e - 8));
    int k = s * 32 + kit;                                 // k = khw*128 + cin
    int n = nt * 16 + (l & 15);
    int khw = k >> 7;
    int cin = k & 127;
    out[tid] = f2bf(w[n * 1152 + cin * 9 + khw]);         // w_off (144,128,3,3)
}

__global__ void swz_wdcn(const float* __restrict__ w, unsigned short* __restrict__ out) {
    int tid = blockIdx.x * blockDim.x + threadIdx.x;      // 18*4*32*16
    int e = tid & 15;
    int l = (tid >> 4) & 31;
    int rest = tid >> 9;
    int nt = rest % 4;
    int s  = rest / 4;                                    // ktile 0..17
    int half = l >> 4;
    int kit = (e < 8) ? (8 * half + e) : (16 + 8 * half + (e - 8));
    int k = s * 32 + kit;                                 // k = cin*9 + kk
    int n = nt * 16 + (l & 15);
    out[tid] = f2bf(w[n * 576 + k]);                      // w_dcn (64,64,3,3)
}

// ===========================================================================
// Pack concat([neig, center]) into zero-padded channel-last bf16 [98*98][128]
// ===========================================================================
__global__ void pack_in(const float* __restrict__ neig, const float* __restrict__ ctr,
                        unsigned short* __restrict__ out) {
    int tid = blockIdx.x * blockDim.x + threadIdx.x;      // 98*98*128
    int c = tid & 127;
    int p = tid >> 7;
    int y = p / PADW, x = p % PADW;
    int h = y - 1, w = x - 1;
    float v = 0.0f;
    if (h >= 0 && h < HH && w >= 0 && w < WW) {
        const float* src = (c < CIN) ? neig : ctr;
        v = src[(c & 63) * HWC + h * WW + w];
    }
    out[tid] = f2bf(v);
}

// ===========================================================================
// GEMM1: offsets[144][9216] = W_off[144][1152] x im2col (implicit, from inpad)
// One wave per (Mtile, 3 consecutive Ntiles). 1728 waves.
// ===========================================================================
__global__ void gemm_offset(const unsigned short* __restrict__ inpad,
                            const unsigned short* __restrict__ wfrag,
                            const float* __restrict__ bias,
                            float* __restrict__ out) {
    int wave = blockIdx.x * (blockDim.x >> 5) + (threadIdx.x >> 5);  // 0..1727
    int lane = threadIdx.x & 31;
    int ntb = (wave % 3) * 3;          // base N tile: 0,3,6
    int mt  = wave / 3;                // 0..575
    int h   = mt / 6;
    int w0  = (mt % 6) * 16;
    int half = lane >> 4;
    int ml   = lane & 15;

    v8f acc0 = {}, acc1 = {}, acc2 = {};
    #pragma unroll 2
    for (int s = 0; s < 36; ++s) {
        int khw = s >> 2;
        int kh = khw / 3, kw = khw % 3;
        int cb = (s & 3) * 32;
        const unsigned short* pa =
            inpad + (size_t)((h + kh) * PADW + (w0 + ml + kw)) * 128 + cb + 8 * half;
        FragBF a, b0, b1, b2;
        a.q[0] = *(const uint4*)(pa);
        a.q[1] = *(const uint4*)(pa + 16);
        const unsigned short* pb = wfrag + (size_t)((s * 9 + ntb) * 32 + lane) * 16;
        b0.q[0] = *(const uint4*)(pb);
        b0.q[1] = *(const uint4*)(pb + 8);
        b1.q[0] = *(const uint4*)(pb + 512);
        b1.q[1] = *(const uint4*)(pb + 520);
        b2.q[0] = *(const uint4*)(pb + 1024);
        b2.q[1] = *(const uint4*)(pb + 1032);
        acc0 = __builtin_amdgcn_wmma_f32_16x16x32_bf16(false, a.v, false, b0.v,
                                                       (short)0, acc0, false, false);
        acc1 = __builtin_amdgcn_wmma_f32_16x16x32_bf16(false, a.v, false, b1.v,
                                                       (short)0, acc1, false, false);
        acc2 = __builtin_amdgcn_wmma_f32_16x16x32_bf16(false, a.v, false, b2.v,
                                                       (short)0, acc2, false, false);
    }
    v8f accs[3] = {acc0, acc1, acc2};
    #pragma unroll
    for (int j = 0; j < 3; ++j) {
        int oc = (ntb + j) * 16 + ml;
        float bv = bias[oc];
        float* po = out + (size_t)oc * HWC + h * WW + w0 + 8 * half;
        float4 v0 = make_float4(accs[j][0] + bv, accs[j][1] + bv,
                                accs[j][2] + bv, accs[j][3] + bv);
        float4 v1 = make_float4(accs[j][4] + bv, accs[j][5] + bv,
                                accs[j][6] + bv, accs[j][7] + bv);
        *(float4*)(po)     = v0;
        *(float4*)(po + 4) = v1;
    }
}

// ===========================================================================
// Bilinear deformable sampling -> bf16 col buffer samp[pos][576],
// k = (g*8+c)*9 + kk  (matches w_dcn cin*9+kk ordering).
// ===========================================================================
__global__ void dcn_sample(const float* __restrict__ feat,    // neig [64][9216]
                           const float* __restrict__ off,     // [144][9216]
                           unsigned short* __restrict__ samp) {
    int tid = blockIdx.x * blockDim.x + threadIdx.x;          // 8*9*9216
    int pos = tid % HWC;
    int gk  = tid / HWC;
    int g = gk / 9, kk = gk % 9;
    int h = pos / WW, w = pos % WW;
    float dy = off[(size_t)(g * 18 + kk * 2 + 0) * HWC + pos];
    float dx = off[(size_t)(g * 18 + kk * 2 + 1) * HWC + pos];
    float ys = (float)(h - 1 + kk / 3) + dy;
    float xs = (float)(w - 1 + kk % 3) + dx;
    float y0f = floorf(ys), x0f = floorf(xs);
    float wy = ys - y0f, wx = xs - x0f;
    int y0 = (int)y0f, x0 = (int)x0f;

    float wgt[4] = {(1.f - wy) * (1.f - wx), (1.f - wy) * wx,
                    wy * (1.f - wx),         wy * wx};
    int yy[4] = {y0, y0, y0 + 1, y0 + 1};
    int xx[4] = {x0, x0 + 1, x0, x0 + 1};
    int   idx[4];
    float msk[4];
    #pragma unroll
    for (int j = 0; j < 4; ++j) {
        bool vok = (yy[j] >= 0) && (yy[j] < HH) && (xx[j] >= 0) && (xx[j] < WW);
        int yc = min(max(yy[j], 0), HH - 1);
        int xc = min(max(xx[j], 0), WW - 1);
        idx[j] = yc * WW + xc;
        msk[j] = vok ? wgt[j] : 0.f;
    }
    unsigned short* po = samp + (size_t)pos * K2 + g * 72 + kk;
    const float* fb = feat + (size_t)(g * 8) * HWC;
    #pragma unroll
    for (int c = 0; c < 8; ++c) {
        const float* fc = fb + (size_t)c * HWC;
        float v = fc[idx[0]] * msk[0] + fc[idx[1]] * msk[1] +
                  fc[idx[2]] * msk[2] + fc[idx[3]] * msk[3];
        po[c * 9] = f2bf(v);
    }
}

// ===========================================================================
// GEMM2: out[64][9216] = W_dcn[64][576] x samp[576][9216]  (+ bias)
// One wave per Mtile, all 4 Ntiles per wave (A reuse x4). 576 waves.
// ===========================================================================
__global__ void gemm_dcn(const unsigned short* __restrict__ samp,
                         const unsigned short* __restrict__ wfrag,
                         const float* __restrict__ bias,
                         float* __restrict__ out) {
    int wave = blockIdx.x * (blockDim.x >> 5) + (threadIdx.x >> 5);  // 0..575
    int lane = threadIdx.x & 31;
    int mt = wave;
    int h  = mt / 6;
    int w0 = (mt % 6) * 16;
    int half = lane >> 4;
    int ml   = lane & 15;

    size_t mrow = (size_t)(h * WW + w0 + ml) * K2;
    v8f acc0 = {}, acc1 = {}, acc2 = {}, acc3 = {};
    #pragma unroll 2
    for (int s = 0; s < 18; ++s) {
        const unsigned short* pa = samp + mrow + s * 32 + 8 * half;
        FragBF a, b0, b1, b2, b3;
        a.q[0] = *(const uint4*)(pa);
        a.q[1] = *(const uint4*)(pa + 16);
        const unsigned short* pb = wfrag + (size_t)((s * 4) * 32 + lane) * 16;
        b0.q[0] = *(const uint4*)(pb);
        b0.q[1] = *(const uint4*)(pb + 8);
        b1.q[0] = *(const uint4*)(pb + 512);
        b1.q[1] = *(const uint4*)(pb + 520);
        b2.q[0] = *(const uint4*)(pb + 1024);
        b2.q[1] = *(const uint4*)(pb + 1032);
        b3.q[0] = *(const uint4*)(pb + 1536);
        b3.q[1] = *(const uint4*)(pb + 1544);
        acc0 = __builtin_amdgcn_wmma_f32_16x16x32_bf16(false, a.v, false, b0.v,
                                                       (short)0, acc0, false, false);
        acc1 = __builtin_amdgcn_wmma_f32_16x16x32_bf16(false, a.v, false, b1.v,
                                                       (short)0, acc1, false, false);
        acc2 = __builtin_amdgcn_wmma_f32_16x16x32_bf16(false, a.v, false, b2.v,
                                                       (short)0, acc2, false, false);
        acc3 = __builtin_amdgcn_wmma_f32_16x16x32_bf16(false, a.v, false, b3.v,
                                                       (short)0, acc3, false, false);
    }
    v8f accs[4] = {acc0, acc1, acc2, acc3};
    #pragma unroll
    for (int j = 0; j < 4; ++j) {
        int oc = j * 16 + ml;
        float bv = bias[oc];
        float* po = out + (size_t)oc * HWC + h * WW + w0 + 8 * half;
        float4 v0 = make_float4(accs[j][0] + bv, accs[j][1] + bv,
                                accs[j][2] + bv, accs[j][3] + bv);
        float4 v1 = make_float4(accs[j][4] + bv, accs[j][5] + bv,
                                accs[j][6] + bv, accs[j][7] + bv);
        *(float4*)(po)     = v0;
        *(float4*)(po + 4) = v1;
    }
}

__global__ void copy_frame(const float* __restrict__ src, float* __restrict__ dst) {
    int i = blockIdx.x * blockDim.x + threadIdx.x;        // 64*9216
    dst[i] = src[i];
}

// ===========================================================================
extern "C" void kernel_launch(void* const* d_in, const int* in_sizes, int n_in,
                              void* d_out, int out_size, void* d_ws, size_t ws_size,
                              hipStream_t stream) {
    const float* x     = (const float*)d_in[0];   // (1,5,64,96,96)
    const float* w_off = (const float*)d_in[1];   // (144,128,3,3)
    const float* b_off = (const float*)d_in[2];   // (144,)
    const float* w_dcn = (const float*)d_in[3];   // (64,64,3,3)
    const float* b_dcn = (const float*)d_in[4];   // (64,)
    float* out = (float*)d_out;

    const size_t FRAME = (size_t)CIN * HWC;       // 589824 floats

    // scratch partition (all 256B aligned)
    char* ws = (char*)d_ws;
    unsigned short* woff_frag = (unsigned short*)(ws);                 //  331776 B
    unsigned short* wdcn_frag = (unsigned short*)(ws + 331776);        //   73728 B
    unsigned short* inpad     = (unsigned short*)(ws + 405504);        // 2458624 B
    float*          off_buf   = (float*)        (ws + 2864128);        // 5308416 B
    unsigned short* samp_buf  = (unsigned short*)(ws + 8172544);       // 10616832 B

    // weight swizzles (once)
    swz_woff<<<648, 256, 0, stream>>>(w_off, woff_frag);               // 165888
    swz_wdcn<<<144, 256, 0, stream>>>(w_dcn, wdcn_frag);               //  36864

    const float* center = x + 2 * FRAME;
    copy_frame<<<2304, 256, 0, stream>>>(center, out + 2 * FRAME);

    const int frames[4] = {0, 1, 3, 4};
    for (int fi = 0; fi < 4; ++fi) {
        int t = frames[fi];
        const float* neig = x + (size_t)t * FRAME;
        float* out_t = out + (size_t)t * FRAME;

        pack_in<<<4802, 256, 0, stream>>>(neig, center, inpad);        // 98*98*128
        gemm_offset<<<432, 128, 0, stream>>>(inpad, woff_frag, b_off, off_buf); // 1728 waves
        dcn_sample<<<2592, 256, 0, stream>>>(neig, off_buf, samp_buf); // 8*9*9216
        gemm_dcn<<<144, 128, 0, stream>>>(samp_buf, wdcn_frag, b_dcn, out_t);   // 576 waves
    }
}
